// AttentionWithBinding_49065706389530
// MI455X (gfx1250) — compile-verified, matrix operand order checked by reading, not run on hardware
//
#include <hip/hip_runtime.h>

// ---------------- problem constants ----------------
#define DIMD      1024
#define NUM_HEADS 16
#define HEAD_DIM  64
#define SEQ       2048
#define BATCH     2
#define ROWS      (BATCH * SEQ)      // 4096
#define SCALE_F   0.125f             // 64^-0.5
#define BIND_F    0.5f

typedef __attribute__((ext_vector_type(16))) __bf16 v16bf;
typedef __attribute__((ext_vector_type(8)))  float  v8f;
typedef unsigned int v4u __attribute__((ext_vector_type(4)));
typedef int          v8i __attribute__((ext_vector_type(8)));
typedef int          v4i __attribute__((ext_vector_type(4)));

// K index packing for 16-bit WMMA operands (ISA 7.12.2):
// lanes 0-15: dword j -> K = {2j,2j+1} (j<4) or {16+2(j-4),...} ; lanes 16-31: +8
__device__ __forceinline__ int wmma_k_of(int j, int half) {
    return ((j < 4) ? (2 * j) : (16 + 2 * (j - 4))) + 8 * half;
}

// A fragment (16x32, M x K), element (row,k) at src[row*stride + k].
// K-pairs contiguous -> 2x ds_load_b128 per fragment.
__device__ __forceinline__ v16bf load_A_frag(const __bf16* src, int stride, int lane) {
    int row = lane & 15, half = lane >> 4;
    v16bf a;
#pragma unroll
    for (int j = 0; j < 8; ++j) {
        int k = wmma_k_of(j, half);
        a[2 * j]     = src[row * stride + k];
        a[2 * j + 1] = src[row * stride + k + 1];
    }
    return a;
}

// B fragment (32x16, K x N) where logical B[k][n] = src[n*stride + k]
// (source stored N x K row-major, i.e. pre-transposed). K-pairs contiguous.
__device__ __forceinline__ v16bf load_B_frag_T(const __bf16* src, int stride, int lane) {
    int col = lane & 15, half = lane >> 4;
    v16bf b;
#pragma unroll
    for (int j = 0; j < 8; ++j) {
        int k = wmma_k_of(j, half);
        b[2 * j]     = src[col * stride + k];
        b[2 * j + 1] = src[col * stride + k + 1];
    }
    return b;
}

__device__ __forceinline__ v8f wmma_bf16(v16bf a, v16bf b, v8f c) {
    return __builtin_amdgcn_wmma_f32_16x16x32_bf16(false, a, false, b, (short)0, c,
                                                   false, false);
}

// ---------------- Tensor Data Mover: 2D tile load (global -> LDS) ----------------
// Packs D# group0/group1 per CDNA5 ISA 8.3/8.4 and issues TENSOR_LOAD_TO_LDS.
// 2D tensor of 2-byte elements: row length tensor_d0 (elems), tensor_d1 rows,
// row stride stride0 (elems). Tile tile_d0 x tile_d1 packed contiguously in LDS.
__device__ __forceinline__ void tdm_load_2d(const void* gaddr, unsigned lds_byte_addr,
                                            unsigned tensor_d0, unsigned tensor_d1,
                                            unsigned tile_d0, unsigned tile_d1,
                                            unsigned stride0) {
    size_t ga = (size_t)gaddr;
    v4u g0;
    g0[0] = 1u;                                   // count=1 valid descriptor
    g0[1] = lds_byte_addr;                        // LDS destination (bytes)
    g0[2] = (unsigned)ga;                         // global_addr[31:0]
    g0[3] = (unsigned)((ga >> 32) & 0x1FFFFFFu)   // global_addr[56:32]
            | (2u << 30);                         // type = 2 (image)
    v8i g1;
    g1[0] = (int)(1u << 16);                      // data_size=1 (2 bytes)
    g1[1] = (int)((tensor_d0 & 0xFFFFu) << 16);   // tensor_dim0[15:0]
    g1[2] = (int)((tensor_d0 >> 16) | ((tensor_d1 & 0xFFFFu) << 16));
    g1[3] = (int)((tensor_d1 >> 16) | (tile_d0 << 16));   // tile_dim0
    g1[4] = (int)(tile_d1 & 0xFFFFu);             // tile_dim1 (tile_dim2 = 0)
    g1[5] = (int)stride0;                         // tensor_dim0_stride[31:0]
    g1[6] = 0;                                    // stride0[47:32], stride1 lo
    g1[7] = 0;
    v4i z4 = {0, 0, 0, 0};
    v8i z8 = {0, 0, 0, 0, 0, 0, 0, 0};
    // 6-arg form (clang-23 / therock headers): groups 0..3 + extra group + cpol
    __builtin_amdgcn_tensor_load_to_lds(g0, g1, z4, z4, z8, 0);
}

__device__ __forceinline__ unsigned lds_off(const void* p) {
    return (unsigned)(size_t)p;  // LDS aperture lives in high bits; low 32 = offset
}

// ---------------- fp32 -> bf16 cast (row-major, for x) ----------------
__global__ __launch_bounds__(256) void cast_f32_to_bf16(const float* __restrict__ src,
                                                        __bf16* __restrict__ dst, int n) {
    int i = (blockIdx.x * 256 + threadIdx.x) * 4;
    if (i + 3 < n) {
        float4 f = *(const float4*)&src[i];
        dst[i]     = (__bf16)f.x;
        dst[i + 1] = (__bf16)f.y;
        dst[i + 2] = (__bf16)f.z;
        dst[i + 3] = (__bf16)f.w;
    }
}

// ---------------- fp32 -> bf16 transposing cast (for weights) ----------------
// dst[c][r] = (bf16) src[r][c];  R x C -> C x R, 32x32 tiles through LDS.
__global__ __launch_bounds__(256) void cast_transpose_bf16(const float* __restrict__ src,
                                                           __bf16* __restrict__ dst,
                                                           int R, int C) {
    __shared__ float tile[32][33];
    int c0 = blockIdx.x * 32, r0 = blockIdx.y * 32;
    int tx = threadIdx.x & 31, ty = threadIdx.x >> 5;  // 32 x 8
#pragma unroll
    for (int i = 0; i < 32; i += 8)
        tile[ty + i][tx] = src[(size_t)(r0 + ty + i) * C + c0 + tx];
    __syncthreads();
#pragma unroll
    for (int i = 0; i < 32; i += 8)
        dst[(size_t)(c0 + ty + i) * R + r0 + tx] = (__bf16)tile[tx][ty + i];
}

// ---------------- bf16 WMMA GEMM (TDM-staged) ----------------
// C[M,N] = (A[M,K] @ W[K,N] + bias) * scale, W supplied pre-transposed as WT[N][K].
// mode 0: bf16 out, [B,H,S,Hd] layout (Q/K projections)
// mode 1: f32 out, row-major (output projection)
// mode 2: bf16 out, [B,H,Hd,S] layout (V projection, pre-transposed for PV GEMM)
__global__ __launch_bounds__(256) void gemm_bf16_kernel(
    const __bf16* __restrict__ A, const __bf16* __restrict__ WT,
    const float* __restrict__ bias, __bf16* __restrict__ out_bf16,
    float* __restrict__ out_f32, int M, int N, int K, int mode, float scale) {
    __shared__ __align__(16) __bf16 As[64][32];    // M x K slab
    __shared__ __align__(16) __bf16 BsT[128][32];  // N x K slab (transposed)

    int tid = threadIdx.x, lane = tid & 31, w = tid >> 5;
    int m0 = blockIdx.y * 64, n0 = blockIdx.x * 128;
    int wm = (w & 3) * 16;        // 4 M-tiles of 16
    int wn = (w >> 2) * 64;       // 2 N-strips of 64 -> 4 tiles each

    v8f acc[4] = {};

    for (int k0 = 0; k0 < K; k0 += 32) {
        if (w == 0) {  // wave 0 drives the Tensor Data Mover for both slabs
            tdm_load_2d(&A[(size_t)m0 * K + k0], lds_off(&As[0][0]),
                        K, M, 32, 64, K);
            tdm_load_2d(&WT[(size_t)n0 * K + k0], lds_off(&BsT[0][0]),
                        K, N, 32, 128, K);
            __builtin_amdgcn_s_wait_tensorcnt((short)0);
        }
        __syncthreads();

        v16bf af = load_A_frag(&As[wm][0], 32, lane);
#pragma unroll
        for (int t = 0; t < 4; ++t) {
            v16bf bf = load_B_frag_T(&BsT[wn + t * 16][0], 32, lane);
            acc[t] = wmma_bf16(af, bf, acc[t]);
        }
        __syncthreads();
    }

    int hf = lane >> 4, cidx = lane & 15;
#pragma unroll
    for (int t = 0; t < 4; ++t) {
#pragma unroll
        for (int r = 0; r < 8; ++r) {
            int gm = m0 + wm + r + 8 * hf;
            int gn = n0 + wn + t * 16 + cidx;
            float val = (acc[t][r] + bias[gn]) * scale;
            if (mode == 1) {
                out_f32[(size_t)gm * N + gn] = val;
            } else {
                int b = gm >> 11, s = gm & (SEQ - 1);       // S = 2048
                int h = gn >> 6, hd = gn & (HEAD_DIM - 1);  // Hd = 64
                if (mode == 0) {
                    out_bf16[(((size_t)(b * NUM_HEADS + h)) * SEQ + s) * HEAD_DIM + hd] =
                        (__bf16)val;
                } else {  // mode 2: V stored [B,H,Hd,S]
                    out_bf16[(((size_t)(b * NUM_HEADS + h)) * HEAD_DIM + hd) * SEQ + s] =
                        (__bf16)val;
                }
            }
        }
    }
}

// ---------------- flash attention with binding bias (TDM-staged) ----------------
// q,k: [B,H,S,Hd] bf16 ; vT: [B,H,Hd,S] bf16
// grid: (B*H, SEQ/128); block: 256 (8 waves); wave w owns q rows [q0, q0+16)
__global__ __launch_bounds__(256) void attn_kernel(
    const __bf16* __restrict__ q, const __bf16* __restrict__ k,
    const __bf16* __restrict__ vT, const float* __restrict__ binding,
    __bf16* __restrict__ aout) {
    __shared__ __align__(16) __bf16 Ks[32][64];     // 32 keys x 64 hd
    __shared__ __align__(16) __bf16 Vt[64][32];     // 64 hd x 32 keys (transposed)
    __shared__ __align__(16) __bf16 Ps[8][16][32];  // per-wave P tile

    int tid = threadIdx.x, lane = tid & 31, w = tid >> 5;
    int hf = lane >> 4, cidx = lane & 15;
    int bh = blockIdx.x;
    int b = bh >> 4, h = bh & 15;
    int q0 = blockIdx.y * 128 + w * 16;

    // Q strip for this wave: two A-fragments (K = 0..31, 32..63), kept in VGPRs
    const __bf16* qbase = q + ((size_t)bh * SEQ + q0) * HEAD_DIM;
    v16bf qa0 = load_A_frag(qbase, HEAD_DIM, lane);
    v16bf qa1 = load_A_frag(qbase + 32, HEAD_DIM, lane);

    v8f o[4] = {};
    float mrow[8], lrow[8];
#pragma unroll
    for (int r = 0; r < 8; ++r) { mrow[r] = -1e30f; lrow[r] = 0.0f; }

    const __bf16* kbp = k + (size_t)bh * SEQ * HEAD_DIM;
    const __bf16* vtp = vT + (size_t)bh * HEAD_DIM * SEQ;

    for (int kk = 0; kk < SEQ; kk += 32) {
        if (w == 0) {  // TDM: K tile (32 keys x 64) and V^T tile (64 hd x 32)
            tdm_load_2d(&kbp[(size_t)kk * HEAD_DIM], lds_off(&Ks[0][0]),
                        HEAD_DIM, SEQ, HEAD_DIM, 32, HEAD_DIM);
            tdm_load_2d(&vtp[kk], lds_off(&Vt[0][0]),
                        SEQ, HEAD_DIM, 32, HEAD_DIM, SEQ);
            __builtin_amdgcn_s_wait_tensorcnt((short)0);
        }
        __syncthreads();

        // scores: S[16 q x 32 keys] = Q(16x64) . K^T ; Q pre-scaled by Hd^-0.5
        v8f s0 = {}, s1 = {};
        s0 = wmma_bf16(qa0, load_B_frag_T(&Ks[0][0], 64, lane), s0);
        s0 = wmma_bf16(qa1, load_B_frag_T(&Ks[0][32], 64, lane), s0);
        s1 = wmma_bf16(qa0, load_B_frag_T(&Ks[16][0], 64, lane), s1);
        s1 = wmma_bf16(qa1, load_B_frag_T(&Ks[16][32], 64, lane), s1);

        // binding bias + online softmax; row r lives in VGPR r for lanes 0-15,
        // row r+8 for lanes 16-31 -> shfl_xor over masks 1,2,4,8 reduces per-row.
#pragma unroll
        for (int r = 0; r < 8; ++r) {
            int grow = q0 + r + 8 * hf;
            float v0 = s0[r] + BIND_F * binding[(size_t)grow * SEQ + kk + cidx];
            float v1 = s1[r] + BIND_F * binding[(size_t)grow * SEQ + kk + 16 + cidx];
            float mt = fmaxf(v0, v1);
            mt = fmaxf(mt, __shfl_xor(mt, 1, 32));
            mt = fmaxf(mt, __shfl_xor(mt, 2, 32));
            mt = fmaxf(mt, __shfl_xor(mt, 4, 32));
            mt = fmaxf(mt, __shfl_xor(mt, 8, 32));
            float mn = fmaxf(mrow[r], mt);
            float corr = __expf(mrow[r] - mn);
            float p0 = __expf(v0 - mn);
            float p1 = __expf(v1 - mn);
            float rs = p0 + p1;
            rs += __shfl_xor(rs, 1, 32);
            rs += __shfl_xor(rs, 2, 32);
            rs += __shfl_xor(rs, 4, 32);
            rs += __shfl_xor(rs, 8, 32);
            lrow[r] = lrow[r] * corr + rs;
            mrow[r] = mn;
            o[0][r] *= corr; o[1][r] *= corr; o[2][r] *= corr; o[3][r] *= corr;
            Ps[w][r + 8 * hf][cidx]      = (__bf16)p0;
            Ps[w][r + 8 * hf][16 + cidx] = (__bf16)p1;
        }

        // O += P(16x32) @ V(32x64); Ps is wave-private, in-wave DS ordering suffices
        v16bf pa = load_A_frag(&Ps[w][0][0], 32, lane);
#pragma unroll
        for (int t = 0; t < 4; ++t) {
            v16bf vf = load_B_frag_T(&Vt[t * 16][0], 32, lane);
            o[t] = wmma_bf16(pa, vf, o[t]);
        }
        __syncthreads();  // protect Ks/Vt before next TDM stage
    }

    // normalize and write to [B, S, D] bf16 for the output projection
#pragma unroll
    for (int r = 0; r < 8; ++r) {
        float inv = 1.0f / lrow[r];
        int s = q0 + r + 8 * hf;
#pragma unroll
        for (int t = 0; t < 4; ++t) {
            aout[((size_t)b * SEQ + s) * DIMD + h * HEAD_DIM + t * 16 + cidx] =
                (__bf16)(o[t][r] * inv);
        }
    }
}

// ---------------- launcher ----------------
extern "C" void kernel_launch(void* const* d_in, const int* in_sizes, int n_in,
                              void* d_out, int out_size, void* d_ws, size_t ws_size,
                              hipStream_t stream) {
    const float* x       = (const float*)d_in[0];
    const float* binding = (const float*)d_in[1];
    const float* Wq = (const float*)d_in[2]; const float* bq = (const float*)d_in[3];
    const float* Wk = (const float*)d_in[4]; const float* bk = (const float*)d_in[5];
    const float* Wv = (const float*)d_in[6]; const float* bv = (const float*)d_in[7];
    const float* Wo = (const float*)d_in[8]; const float* bo = (const float*)d_in[9];
    float* out = (float*)d_out;

    char* ws = (char*)d_ws;
    const size_t N_X = (size_t)ROWS * DIMD;      // 4,194,304
    const size_t N_W = (size_t)DIMD * DIMD;      // 1,048,576
    __bf16* xb  = (__bf16*)(ws);
    __bf16* wqt = (__bf16*)(ws + N_X * 2);       // W^T, bf16, [N][K]
    __bf16* wkt = wqt + N_W;
    __bf16* wvt = wkt + N_W;
    __bf16* wot = wvt + N_W;
    __bf16* qb  = wot + N_W;                     // [B,H,S,Hd]
    __bf16* kb  = qb + N_X;                      // [B,H,S,Hd]
    __bf16* vbt = kb + N_X;                      // [B,H,Hd,S]
    __bf16* ab  = vbt + N_X;                     // [B,S,D]

    cast_f32_to_bf16<<<(int)(N_X / 4 / 256), 256, 0, stream>>>(x, xb, (int)N_X);
    dim3 tgrid(DIMD / 32, DIMD / 32);            // (32, 32)
    cast_transpose_bf16<<<tgrid, 256, 0, stream>>>(Wq, wqt, DIMD, DIMD);
    cast_transpose_bf16<<<tgrid, 256, 0, stream>>>(Wk, wkt, DIMD, DIMD);
    cast_transpose_bf16<<<tgrid, 256, 0, stream>>>(Wv, wvt, DIMD, DIMD);
    cast_transpose_bf16<<<tgrid, 256, 0, stream>>>(Wo, wot, DIMD, DIMD);

    dim3 ggrid(DIMD / 128, ROWS / 64);  // (8, 64)
    gemm_bf16_kernel<<<ggrid, 256, 0, stream>>>(xb, wqt, bq, qb, nullptr,
                                                ROWS, DIMD, DIMD, 0, SCALE_F);
    gemm_bf16_kernel<<<ggrid, 256, 0, stream>>>(xb, wkt, bk, kb, nullptr,
                                                ROWS, DIMD, DIMD, 0, 1.0f);
    gemm_bf16_kernel<<<ggrid, 256, 0, stream>>>(xb, wvt, bv, vbt, nullptr,
                                                ROWS, DIMD, DIMD, 2, 1.0f);

    dim3 agrid(BATCH * NUM_HEADS, SEQ / 128);   // (32, 16)
    attn_kernel<<<agrid, 256, 0, stream>>>(qb, kb, vbt, binding, ab);

    gemm_bf16_kernel<<<ggrid, 256, 0, stream>>>(ab, wot, bo, nullptr, out,
                                                ROWS, DIMD, DIMD, 1, 1.0f);
}
